// PSModule_62397284876666
// MI455X (gfx1250) — matrix operand order repaired and checked by reading, not compiled, not run on hardware
//
#include <hip/hip_runtime.h>

// ---------------------------------------------------------------------------
// MI455X (gfx1250) fused deconv(2048->1024, k3 s2 p1 op1) + ReLU + 1x1(->8)
// + class-balanced CE, via bf16 WMMA with hi/lo fp32 compensation.
//   D(16x16,f32) = A(16x32,bf16) x B(32x16,bf16) + C   (v_wmma_f32_16x16x32_bf16)
// ---------------------------------------------------------------------------

typedef __attribute__((ext_vector_type(16))) __bf16 v16bf;
typedef __attribute__((ext_vector_type(8)))  float  v8f;

union FragU { uint4 u[2]; v16bf v; };

__device__ inline v8f wmma_bf16(v16bf a, v16bf b, v8f c) {
  return __builtin_amdgcn_wmma_f32_16x16x32_bf16(
      /*neg_a=*/false, a, /*neg_b=*/false, b,
      /*c_mod=*/(short)0, c, /*reuse_a=*/false, /*reuse_b=*/false);
}

__device__ inline unsigned short cvt_bf16(float f) {
  unsigned u = __float_as_uint(f);
  if ((u & 0x7fffffffu) > 0x7f800000u) return (unsigned short)((u >> 16) | 0x40u);
  return (unsigned short)((u + 0x7fffu + ((u >> 16) & 1u)) >> 16);
}
__device__ inline float bf16_hi_f32(unsigned short h) {
  return __uint_as_float(((unsigned)h) << 16);
}

// ---------------------------------------------------------------------------
// prep_w: (c=2048, m=1024, ky, kx) fp32  ->  Whi/Wlo[tap][m][c] bf16
// Tiled through LDS: per block, 32 c-rows x (32 m x 9 taps)=288 contiguous
// floats in, contiguous-c bf16 rows out.
// ---------------------------------------------------------------------------
__global__ __launch_bounds__(256)
void prep_w(const float* __restrict__ w, unsigned short* __restrict__ whi,
            unsigned short* __restrict__ wlo) {
  __shared__ float s[32 * 288];
  const int m0 = blockIdx.x * 32;   // 1024/32 = 32 blocks
  const int c0 = blockIdx.y * 32;   // 2048/32 = 64 blocks
  const int tid = threadIdx.x;
#pragma unroll 1
  for (int it = 0; it < 36; ++it) {
    int f = tid + it * 256;
    int cl = f / 288, e = f - cl * 288;           // e = m_local*9 + tap
    s[cl * 288 + e] = w[(size_t)(c0 + cl) * 9216 + (size_t)m0 * 9 + e];
  }
  __syncthreads();
#pragma unroll 1
  for (int it = 0; it < 36; ++it) {
    int g = tid + it * 256;                        // 9 taps * 32m * 32c
    int tap = g >> 10, mm = (g >> 5) & 31, cc = g & 31;
    float v = s[cc * 288 + mm * 9 + tap];
    unsigned short h = cvt_bf16(v);
    float lo = v - bf16_hi_f32(h);
    size_t o = ((size_t)tap * 1024 + (m0 + mm)) * 2048 + (c0 + cc);
    whi[o] = h;
    wlo[o] = cvt_bf16(lo);
  }
}

// ---------------------------------------------------------------------------
// prep_f: (b, c=2048, y, x) fp32 NCHW -> Fhi/Flo[b][y][x][c] bf16 (NHWC)
// ---------------------------------------------------------------------------
__global__ __launch_bounds__(256)
void prep_f(const float* __restrict__ feat, unsigned short* __restrict__ fhi,
            unsigned short* __restrict__ flo) {
  __shared__ float s[32 * 33];
  const int yx0 = blockIdx.x * 32;  // 576/32 = 18 blocks
  const int c0 = blockIdx.y * 32;   // 64 blocks
  const int b = blockIdx.z;         // 4
  const int tid = threadIdx.x;
#pragma unroll
  for (int it = 0; it < 4; ++it) {
    int f = tid + it * 256;
    int cl = f >> 5, xl = f & 31;
    s[cl * 33 + xl] = feat[((size_t)b * 2048 + c0 + cl) * 576 + yx0 + xl];
  }
  __syncthreads();
#pragma unroll
  for (int it = 0; it < 4; ++it) {
    int f = tid + it * 256;
    int xl = f >> 5, cl = f & 31;
    float v = s[cl * 33 + xl];
    unsigned short h = cvt_bf16(v);
    float lo = v - bf16_hi_f32(h);
    size_t o = ((size_t)b * 576 + yx0 + xl) * 2048 + (c0 + cl);
    fhi[o] = h;
    flo[o] = cvt_bf16(lo);
  }
}

// ---------------------------------------------------------------------------
// Fused main kernel. Grid (72 pixel-tiles, 4 phases), 256 threads (8 waves).
// Phase (dy,dx): output pixel (2*py+dy, 2*px+dx); taps per dim:
//   d==0 -> {k=1, off=0};  d==1 -> {k=2, off=0}, {k=0, off=+1 (bounds-checked)}
// Wave w owns M rows [w*128, w*128+128) for the WG's 32 pixels, K = 2048.
// ---------------------------------------------------------------------------
__global__ __launch_bounds__(256)
void deconv_head(const unsigned short* __restrict__ Whi,
                 const unsigned short* __restrict__ Wlo,
                 const unsigned short* __restrict__ Fhi,
                 const unsigned short* __restrict__ Flo,
                 const float* __restrict__ bdec,
                 const float* __restrict__ wconv,
                 const float* __restrict__ bconv,
                 const int* __restrict__ labels,
                 float* __restrict__ part_out) {
  __shared__ float s_wconv[8 * 1024];   // 32 KB
  __shared__ float s_bias[1024];        // 4 KB
  __shared__ float s_part[16][32][9];   // 18 KB (pad 9 to dodge bank conflicts)
  __shared__ float s_logits[32][8];
  __shared__ float s_ce[32];
  __shared__ int s_lbl[32];

  const int tid = threadIdx.x;
  const int lane = tid & 31;
  const int lane16 = lane & 15;
  const int hihalf = lane >> 4;
  const int w = tid >> 5;
  const int mBase = w * 128;
  const int phase = blockIdx.y;
  const int dy = phase >> 1, dx = phase & 1;
  const int q0 = blockIdx.x * 32;

  for (int it = tid; it < 8192; it += 256) s_wconv[it] = wconv[it];
  for (int it = tid; it < 1024; it += 256) s_bias[it] = bdec[it];
  __syncthreads();

  // Per N-subtile pixel decomposition (lane -> pixel).
  int pb[2], py[2], px[2];
#pragma unroll
  for (int j = 0; j < 2; ++j) {
    int q = q0 + j * 16 + lane16;
    pb[j] = q / 576;
    int r = q - pb[j] * 576;
    py[j] = r / 24;
    px[j] = r - py[j] * 24;
  }
  const int khalf = hihalf * 16;  // B frag: lanes 0-15 hold K 0..15, 16-31 hold K 16..31
  const int ka0 = hihalf ? 16 : 0; // A frag: lanes 0-15 K 0..15; 16-31 K 16..31

  v8f acc[8][2];
#pragma unroll
  for (int i = 0; i < 8; ++i) {
    v8f z = {};
    acc[i][0] = z;
    acc[i][1] = z;
  }

  const int nty = dy ? 2 : 1, ntx = dx ? 2 : 1;
  const int ntaps = nty * ntx;

#pragma unroll 1
  for (int t = 0; t < ntaps; ++t) {
    const int ty = t / ntx, tx = t - ty * ntx;
    const int oy_off = (dy && ty) ? 1 : 0;
    const int ky = (dy == 0) ? 1 : (ty ? 0 : 2);
    const int ox_off = (dx && tx) ? 1 : 0;
    const int kx = (dx == 0) ? 1 : (tx ? 0 : 2);
    const int tap = ky * 3 + kx;

    size_t sb[2];
    bool vld[2];
#pragma unroll
    for (int j = 0; j < 2; ++j) {
      int iy = py[j] + oy_off, ix = px[j] + ox_off;
      vld[j] = (iy < 24) && (ix < 24);
      int iyc = vld[j] ? iy : 0, ixc = vld[j] ? ix : 0;
      sb[j] = ((size_t)((pb[j] * 24 + iyc) * 24 + ixc)) * 2048 + khalf;
    }
    const unsigned short* wh = Whi + (size_t)tap * 1024 * 2048;
    const unsigned short* wl = Wlo + (size_t)tap * 1024 * 2048;

#pragma unroll 1
    for (int kt = 0; kt < 64; ++kt) {
      const int k0 = kt * 32;
      // Prefetch next K-tile of this wave's weight slab (global_prefetch_b8).
      __builtin_prefetch((const void*)(wh + (size_t)mBase * 2048 + (size_t)(k0 + 32)), 0, 0);

      FragU bh[2], bl[2];
#pragma unroll
      for (int j = 0; j < 2; ++j) {
        if (vld[j]) {
          const uint4* p = (const uint4*)(Fhi + sb[j] + k0);
          bh[j].u[0] = p[0];
          bh[j].u[1] = p[1];
          const uint4* p2 = (const uint4*)(Flo + sb[j] + k0);
          bl[j].u[0] = p2[0];
          bl[j].u[1] = p2[1];
        } else {
          bh[j].u[0] = make_uint4(0, 0, 0, 0);
          bh[j].u[1] = make_uint4(0, 0, 0, 0);
          bl[j].u[0] = make_uint4(0, 0, 0, 0);
          bl[j].u[1] = make_uint4(0, 0, 0, 0);
        }
      }
#pragma unroll
      for (int i = 0; i < 8; ++i) {
        const size_t arow =
            ((size_t)(mBase + i * 16 + lane16)) * 2048 + (size_t)(k0 + ka0);
        FragU ah, al;
        ah.u[0] = *(const uint4*)(wh + arow);
        ah.u[1] = *(const uint4*)(wh + arow + 8);
        al.u[0] = *(const uint4*)(wl + arow);
        al.u[1] = *(const uint4*)(wl + arow + 8);
#pragma unroll
        for (int j = 0; j < 2; ++j) {
          acc[i][j] = wmma_bf16(ah.v, bh[j].v, acc[i][j]);  // hi*hi
          acc[i][j] = wmma_bf16(ah.v, bl[j].v, acc[i][j]);  // hi*lo
          acc[i][j] = wmma_bf16(al.v, bh[j].v, acc[i][j]);  // lo*hi
        }
      }
    }
  }

  // Epilogue: bias + ReLU + per-lane partial 1x1-conv logits.
  // D layout: lane L, VGPR r of tile (i,j): m = mBase+i*16+r+8*(L>=16), pixel = j*16+(L&15)
  const int src = (w << 1) | hihalf;
#pragma unroll
  for (int j = 0; j < 2; ++j) {
    float p8[8];
#pragma unroll
    for (int k = 0; k < 8; ++k) p8[k] = 0.f;
#pragma unroll
    for (int i = 0; i < 8; ++i) {
#pragma unroll
      for (int r = 0; r < 8; ++r) {
        int m = mBase + i * 16 + r + (hihalf << 3);
        float x = fmaxf(acc[i][j][r] + s_bias[m], 0.f);
#pragma unroll
        for (int k = 0; k < 8; ++k) p8[k] = fmaf(x, s_wconv[k * 1024 + m], p8[k]);
      }
    }
    const int pixL = j * 16 + lane16;
#pragma unroll
    for (int k = 0; k < 8; ++k) s_part[src][pixL][k] = p8[k];
  }
  __syncthreads();

  // Combine 16 partial sources per (pixel, class).
  {
    int pix = tid >> 3, k = tid & 7;
    float lg = bconv[k];
#pragma unroll
    for (int s2 = 0; s2 < 16; ++s2) lg += s_part[s2][pix][k];
    s_logits[pix][k] = lg;
  }
  __syncthreads();

  // Per-pixel cross entropy.
  if (tid < 32) {
    int q = q0 + tid;
    int b = q / 576;
    int r = q - b * 576;
    int pyy = r / 24;
    int pxx = r - pyy * 24;
    int oy = 2 * pyy + dy, ox = 2 * pxx + dx;
    int lbl = labels[(b * 48 + oy) * 48 + ox];
    float mx = s_logits[tid][0];
#pragma unroll
    for (int k = 1; k < 8; ++k) mx = fmaxf(mx, s_logits[tid][k]);
    float se = 0.f;
#pragma unroll
    for (int k = 0; k < 8; ++k) se += expf(s_logits[tid][k] - mx);
    s_ce[tid] = mx + logf(se) - s_logits[tid][lbl];
    s_lbl[tid] = lbl;
  }
  __syncthreads();

  // Deterministic per-WG per-class sums/counts (no float atomics).
  if (tid < 8) {
    float sum = 0.f, cnt = 0.f;
    for (int p = 0; p < 32; ++p) {
      if (s_lbl[p] == tid) {
        sum += s_ce[p];
        cnt += 1.f;
      }
    }
    const int wg = blockIdx.y * 72 + blockIdx.x;
    part_out[wg * 16 + tid] = sum;
    part_out[wg * 16 + 8 + tid] = cnt;
  }
}

// ---------------------------------------------------------------------------
// finalize: reduce 288 WG partials -> class-balanced CE loss scalar.
// ---------------------------------------------------------------------------
__global__ __launch_bounds__(64)
void finalize(const float* __restrict__ part, float* __restrict__ out) {
  __shared__ float s[16];
  const int t = threadIdx.x;
  if (t < 16) {
    float a = 0.f;
    for (int wg = 0; wg < 288; ++wg) a += part[wg * 16 + t];
    s[t] = a;
  }
  __syncthreads();
  if (t == 0) {
    float ls = 0.f, np = 0.f;
    for (int k = 0; k < 8; ++k) {
      float c = s[8 + k];
      if (c > 0.f) {
        ls += s[k] / c;
        np += 1.f;
      }
    }
    out[0] = ls / fmaxf(np, 1.f);
  }
}

// ---------------------------------------------------------------------------
extern "C" void kernel_launch(void* const* d_in, const int* in_sizes, int n_in,
                              void* d_out, int out_size, void* d_ws, size_t ws_size,
                              hipStream_t stream) {
  (void)in_sizes; (void)n_in; (void)out_size; (void)ws_size;
  const float* feat = (const float*)d_in[0];   // (4,2048,24,24)
  const float* wdec = (const float*)d_in[1];   // (2048,1024,3,3)
  const float* bdec = (const float*)d_in[2];   // (1024,)
  const float* wcv  = (const float*)d_in[3];   // (8,1024)
  const float* bcv  = (const float*)d_in[4];   // (8,)
  const int*   lbl  = (const int*)d_in[5];     // (4,48,48) int32

  char* ws = (char*)d_ws;
  unsigned short* whi = (unsigned short*)(ws);                 // 37,748,736 B
  unsigned short* wlo = (unsigned short*)(ws + 37748736);      // 37,748,736 B
  unsigned short* fhi = (unsigned short*)(ws + 75497472);      //  9,437,184 B
  unsigned short* flo = (unsigned short*)(ws + 84934656);      //  9,437,184 B
  float*          prt = (float*)(ws + 94371840);               // 288*16 floats

  prep_w<<<dim3(32, 64), 256, 0, stream>>>(wdec, whi, wlo);
  prep_f<<<dim3(18, 64, 4), 256, 0, stream>>>(feat, fhi, flo);
  deconv_head<<<dim3(72, 4), 256, 0, stream>>>(whi, wlo, fhi, flo, bdec, wcv,
                                               bcv, lbl, prt);
  finalize<<<1, 64, 0, stream>>>(prt, (float*)d_out);
}
